// GAT_14422500180358
// MI455X (gfx1250) — matrix operand order repaired and checked by reading, not compiled, not run on hardware
//
#include <hip/hip_runtime.h>
#include <hip/hip_bf16.h>

// ---------------------------------------------------------------------------
// GAT on MI455X (gfx1250): bf16 WMMA for both GEMMs, f32 softmax, 2-pass
// flash-style attention. LDS staging of shared B-operands (W[h], hT[b,h])
// with bank-conflict padding; double-buffered fragments to overlap
// loads / VALU softmax math with v_wmma issue.
// ---------------------------------------------------------------------------

typedef __attribute__((ext_vector_type(16))) __bf16 v16bf;
typedef __attribute__((ext_vector_type(8)))  float  v8f;

#define GAT_ALPHA    0.2f
#define GAT_MASKVAL  -9.0e15f

struct alignas(16) U4 { unsigned int x, y, z, w; };   // POD 16B for b128 moves

union BF16Frag { v16bf v; unsigned int u[8]; U4 q[2]; };
union AccFrag  { v8f  v; float f[8]; };

__device__ __forceinline__ unsigned int f32_to_bf16(float f) {
    unsigned int u = __float_as_uint(f);
    unsigned int r = ((u >> 16) & 1u) + 0x7FFFu;   // round-to-nearest-even
    return (u + r) >> 16;
}
__device__ __forceinline__ unsigned int pack_bf16(float lo, float hi) {
    return f32_to_bf16(lo) | (f32_to_bf16(hi) << 16);
}
__device__ __forceinline__ float bf16_to_f32(unsigned short h) {
    return __uint_as_float(((unsigned int)h) << 16);
}

// --------------------------- f32 -> bf16 convert ---------------------------
__global__ void gat_cvt_bf16(const float* __restrict__ src,
                             unsigned short* __restrict__ dst, int n) {
    int i = blockIdx.x * blockDim.x + threadIdx.x;
    if (i < n) dst[i] = (unsigned short)f32_to_bf16(src[i]);
}

// ------------------- GEMM1: hT[b,h,k,n] = sum_d x[b,n,d] W[h,k,d] ----------
// Block = one (b,h) and 64 rows of n; waves own 16-row tiles. W[h] (32KB)
// and the x tile (32KB) are staged in LDS (rows padded by 8 bf16 to spread
// banks). Fragments are double-buffered across the d-loop.
__global__ __launch_bounds__(128) void gat_gemm1(
    const unsigned short* __restrict__ xbf,   // [B*N, D] bf16
    const unsigned short* __restrict__ wbf,   // [H, K, D] bf16
    unsigned short* __restrict__ hT,          // [B*H, K, N] bf16
    int B, int N, int D, int H, int K)
{
    extern __shared__ unsigned short smem[];
    const int DP = 256 + 8;                       // padded row length (D=256)
    unsigned short* ldsW = smem;                  // [64][DP]
    unsigned short* ldsX = smem + 64 * DP;        // [64][DP]

    const int tid   = threadIdx.x;
    const int lane  = tid & 31;
    const int wave  = tid >> 5;
    const int li    = lane & 15;
    const int base  = (lane < 16) ? 0 : 8;        // A-frag K split / C-row split
    const int hlf16 = (lane < 16) ? 0 : 16;       // unused (B-frag K split is base)
    const int mt    = blockIdx.x % (N / 64);
    const int bh    = blockIdx.x / (N / 64);
    const int h     = bh % H;
    const int b     = bh / H;
    const int n0    = mt * 64 + wave * 16;

    // ---- Stage W[h] and x tile into LDS (coalesced b128 copies) ----
    {
        const int r0  = tid >> 5;                 // 0..3
        const int col = tid & 31;                 // 32 x U4 = 256 bf16 per row
        const unsigned short* wsrc = wbf + (size_t)h * K * D;
        const unsigned short* xsrc = xbf + ((size_t)b * N + (size_t)mt * 64) * D;
        for (int r = r0; r < 64; r += 4) {
            ((U4*)(ldsW + r * DP))[col] = ((const U4*)(wsrc + (size_t)r * D))[col];
            ((U4*)(ldsX + r * DP))[col] = ((const U4*)(xsrc + (size_t)r * D))[col];
        }
    }
    __syncthreads();

    AccFrag acc[4];
#pragma unroll
    for (int t = 0; t < 4; ++t)
#pragma unroll
        for (int r = 0; r < 8; ++r) acc[t].f[r] = 0.0f;

    const unsigned short* xrow = ldsX + (wave * 16 + li) * DP;

    auto loadA = [&](BF16Frag& fr, int d0) {
        fr.q[0] = *(const U4*)(xrow + d0 + base);
        fr.q[1] = *(const U4*)(xrow + d0 + 16 + base);
    };
    auto loadB = [&](BF16Frag& fr, int t, int d0) {
        const unsigned short* wr = ldsW + (t * 16 + li) * DP + d0 + base;
        fr.q[0] = *(const U4*)(wr);
        fr.q[1] = *(const U4*)(wr + 16);
    };

    BF16Frag a, bb[4];
    loadA(a, 0);
#pragma unroll
    for (int t = 0; t < 4; ++t) loadB(bb[t], t, 0);

#pragma unroll
    for (int it = 0; it < 8; ++it) {              // D/32 = 8 steps
        const int d0 = it * 32;
        BF16Frag an = a, bn[4];
#pragma unroll
        for (int t = 0; t < 4; ++t) bn[t] = bb[t];
        if (it + 1 < 8) {
            loadA(an, d0 + 32);
#pragma unroll
            for (int t = 0; t < 4; ++t) loadB(bn[t], t, d0 + 32);
        }
#pragma unroll
        for (int t = 0; t < 4; ++t)
            acc[t].v = __builtin_amdgcn_wmma_f32_16x16x32_bf16(
                false, a.v, false, bb[t].v, (short)0, acc[t].v, false, false);
        a = an;
#pragma unroll
        for (int t = 0; t < 4; ++t) bb[t] = bn[t];
    }

    // Store transposed (k-major): for fixed (t, lane) rows are consecutive n.
#pragma unroll
    for (int t = 0; t < 4; ++t) {
        const int kcol = t * 16 + li;
        unsigned short* dst = hT + ((size_t)bh * K + kcol) * N + n0 + base;
#pragma unroll
        for (int q = 0; q < 4; ++q) {
            unsigned int pr = pack_bf16(acc[t].f[2 * q], acc[t].f[2 * q + 1]);
            *reinterpret_cast<unsigned int*>(dst + 2 * q) = pr;
        }
    }
}

// ----------------- s_self/s_nb = h . a[:K] / a[K:] per (b,h,n) -------------
__global__ void gat_attn_vec(const unsigned short* __restrict__ hT,
                             const float* __restrict__ a,
                             float* __restrict__ s_self,
                             float* __restrict__ s_nb,
                             int B, int N, int H, int K)
{
    int idx = blockIdx.x * blockDim.x + threadIdx.x;     // (b*H+h)*N + n
    int total = B * H * N;
    if (idx >= total) return;
    int n  = idx % N;
    int bh = idx / N;
    int h  = bh % H;
    const float* a1 = a + (size_t)h * 2 * K;
    const float* a2 = a1 + K;
    float ss = 0.0f, sn = 0.0f;
    for (int k = 0; k < K; ++k) {
        float hv = bf16_to_f32(hT[((size_t)bh * K + k) * N + n]);
        ss = fmaf(hv, a1[k], ss);
        sn = fmaf(hv, a2[k], sn);
    }
    s_self[idx] = ss;
    s_nb[idx]   = sn;
}

// --------------- pass 1: per-row online softmax max & sum ------------------
__global__ void gat_softmax_stats(const float* __restrict__ s_self,
                                  const float* __restrict__ s_nb,
                                  const int* __restrict__ adj,
                                  float* __restrict__ mrow,
                                  float* __restrict__ lrow,
                                  int B, int N, int H)
{
    int idx = blockIdx.x * blockDim.x + threadIdx.x;     // (b*H+h)*N + i
    int total = B * H * N;
    if (idx >= total) return;
    int i  = idx % N;
    int bh = idx / N;
    int b  = bh / H;
    const float si   = s_self[idx];
    const float* snb = s_nb + (size_t)bh * N;
    const int* arow  = adj + ((size_t)b * N + i) * N;
    float m = -3.0e38f, l = 0.0f;
    for (int j = 0; j < N; ++j) {
        if ((j & 31) == 0) __builtin_prefetch(arow + j + 128, 0, 0);
        float e = si + snb[j];
        e = (e > 0.0f) ? e : GAT_ALPHA * e;
        e = (arow[j] != 0) ? e : GAT_MASKVAL;
        float mn = fmaxf(m, e);
        l = l * __expf(m - mn) + __expf(e - mn);
        m = mn;
    }
    mrow[idx] = m;
    lrow[idx] = l;
}

// -------- pass 2: out = elu( softmax(e) . h ), fused via WMMA --------------
// Block = one (b,h), 64 rows of i. The whole hT[b,h] (64x1024 bf16, padded)
// plus the s_nb row is staged in LDS once; the j-loop builds P tiles in the
// A-fragment layout (double-buffered so exp VALU overlaps WMMA issue).
__global__ __launch_bounds__(128) void gat_attn_out(
    const unsigned short* __restrict__ hT,   // [B*H, K, N] bf16
    const float* __restrict__ s_self,
    const float* __restrict__ s_nb,
    const int* __restrict__ adj,
    const float* __restrict__ mrow,
    const float* __restrict__ lrow,
    float* __restrict__ out,                 // [B, N, H*K] f32
    int B, int N, int H, int K)
{
    extern __shared__ unsigned short smem[];
    const int NP = 1024 + 8;                      // padded row length (N=1024)
    unsigned short* ldsH = smem;                  // [64][NP]
    float* ldsSnb = (float*)(smem + 64 * NP);     // [N]

    const int tid   = threadIdx.x;
    const int lane  = tid & 31;
    const int wave  = tid >> 5;
    const int li    = lane & 15;
    const int half8 = (lane < 16) ? 0 : 8;
    const int hlf16 = (lane < 16) ? 0 : 16;
    const int it    = blockIdx.x % (N / 64);
    const int bh    = blockIdx.x / (N / 64);
    const int b     = bh / H;
    const int h     = bh % H;
    const int i0    = it * 64 + wave * 16;
    const int irow  = i0 + li;

    // ---- Stage hT[b,h] and s_nb[b,h] into LDS ----
    {
        const unsigned short* hsrc = hT + (size_t)bh * K * N;
        for (int k = 0; k < 64; ++k)              // 128 x U4 = 1024 bf16 per row
            ((U4*)(ldsH + k * NP))[tid] = ((const U4*)(hsrc + (size_t)k * N))[tid];
        const float* snbsrc = s_nb + (size_t)bh * N;
        for (int i = tid; i < 256; i += 128)      // 1024 f32 = 256 x U4
            ((U4*)ldsSnb)[i] = ((const U4*)snbsrc)[i];
    }
    __syncthreads();

    const float si = s_self[(size_t)bh * N + irow];
    const float mi = mrow[(size_t)bh * N + irow];
    const int* arow = adj + ((size_t)b * N + irow) * N;

    AccFrag acc[4];
#pragma unroll
    for (int t = 0; t < 4; ++t)
#pragma unroll
        for (int r = 0; r < 8; ++r) acc[t].f[r] = 0.0f;

    auto computeP = [&](BF16Frag& pa, int j0) {
        __builtin_prefetch(arow + j0 + 64, 0, 0);
#pragma unroll
        for (int v = 0; v < 8; ++v) {
            int jb = j0 + ((v < 4) ? (half8 + 2 * v) : (16 + half8 + 2 * (v - 4)));
            float e0 = si + ldsSnb[jb];
            e0 = (e0 > 0.0f) ? e0 : GAT_ALPHA * e0;
            float p0 = (arow[jb] != 0) ? __expf(e0 - mi) : 0.0f;
            float e1 = si + ldsSnb[jb + 1];
            e1 = (e1 > 0.0f) ? e1 : GAT_ALPHA * e1;
            float p1 = (arow[jb + 1] != 0) ? __expf(e1 - mi) : 0.0f;
            pa.u[v] = pack_bf16(p0, p1);
        }
    };

    BF16Frag pcur;
    computeP(pcur, 0);

    for (int j0 = 0; j0 < N; j0 += 32) {
        BF16Frag bb[4];
#pragma unroll
        for (int t = 0; t < 4; ++t) {
            const unsigned short* hb = ldsH + (t * 16 + li) * NP + j0 + half8;
            bb[t].q[0] = *(const U4*)(hb);
            bb[t].q[1] = *(const U4*)(hb + 16);
        }
        BF16Frag pnext = pcur;
        if (j0 + 32 < N) computeP(pnext, j0 + 32);   // VALU overlaps WMMA below
#pragma unroll
        for (int t = 0; t < 4; ++t)
            acc[t].v = __builtin_amdgcn_wmma_f32_16x16x32_bf16(
                false, pcur.v, false, bb[t].v, (short)0, acc[t].v, false, false);
        pcur = pnext;
    }

    // Epilogue: scale rows by 1/l, ELU, write merged-cat output [B,N,H*K].
#pragma unroll
    for (int r = 0; r < 8; ++r) {
        const int rowi = i0 + half8 + r;
        const float inv_l = 1.0f / lrow[(size_t)bh * N + rowi];
#pragma unroll
        for (int t = 0; t < 4; ++t) {
            float v = acc[t].f[r] * inv_l;
            v = (v > 0.0f) ? v : (__expf(v) - 1.0f);   // elu, alpha=1
            const int k = t * 16 + li;
            out[(((size_t)b * N + rowi) * H + h) * K + k] = v;
        }
    }
}

// ---------------------------------------------------------------------------
extern "C" void kernel_launch(void* const* d_in, const int* in_sizes, int n_in,
                              void* d_out, int out_size, void* d_ws, size_t ws_size,
                              hipStream_t stream) {
    (void)in_sizes; (void)n_in; (void)out_size; (void)ws_size;
    const float* x   = (const float*)d_in[0];   // [B,N,D]
    const int*   adj = (const int*)d_in[1];     // [B,N,N]
    const float* W   = (const float*)d_in[2];   // [H,K,D]
    const float* a   = (const float*)d_in[3];   // [H,2K]
    float* out = (float*)d_out;                 // [B,N,H*K]

    const int B = 8, N = 1024, D = 256, H = 8, K = 64;

    char* ws = (char*)d_ws;
    size_t off = 0;
    auto alloc = [&](size_t bytes) -> void* {
        void* p = ws + off;
        off = (off + bytes + 255) & ~(size_t)255;
        return p;
    };
    unsigned short* xbf  = (unsigned short*)alloc((size_t)B * N * D * 2);
    unsigned short* wbf  = (unsigned short*)alloc((size_t)H * K * D * 2);
    unsigned short* hT   = (unsigned short*)alloc((size_t)B * H * K * N * 2);
    float* s_self = (float*)alloc((size_t)B * H * N * 4);
    float* s_nb   = (float*)alloc((size_t)B * H * N * 4);
    float* mr     = (float*)alloc((size_t)B * H * N * 4);
    float* lr     = (float*)alloc((size_t)B * H * N * 4);

    const int nx = B * N * D;
    const int nw = H * K * D;
    gat_cvt_bf16<<<(nx + 255) / 256, 256, 0, stream>>>(x, xbf, nx);
    gat_cvt_bf16<<<(nw + 255) / 256, 256, 0, stream>>>(W, wbf, nw);

    const size_t smem_g1 = (size_t)2 * 64 * (256 + 8) * 2;            // 66 KB
    gat_gemm1<<<B * H * (N / 64), 128, smem_g1, stream>>>(xbf, wbf, hT,
                                                          B, N, D, H, K);

    const int rows = B * H * N;
    gat_attn_vec<<<(rows + 255) / 256, 256, 0, stream>>>(hT, a, s_self, s_nb,
                                                         B, N, H, K);
    gat_softmax_stats<<<(rows + 255) / 256, 256, 0, stream>>>(s_self, s_nb, adj,
                                                              mr, lr, B, N, H);

    const size_t smem_at = (size_t)64 * (1024 + 8) * 2 + (size_t)N * 4; // ~133 KB
    gat_attn_out<<<B * H * (N / 64), 128, smem_at, stream>>>(
        hT, s_self, s_nb, adj, mr, lr, out, B, N, H, K);
}
